// BatchAligneToReference_2190433320998
// MI455X (gfx1250) — matrix operand order repaired and checked by reading, not compiled, not run on hardware
//
#include <hip/hip_runtime.h>

// Batched circular cross-correlation + argmax + circular alignment for
// B=32, S=64, N=8192  (2048 independent rows).
// corr[k] = sum_n x[n]*xref[(n+k) mod N]  computed as Toeplitz GEMM with
// v_wmma_f32_16x16x32_bf16 (f32 accumulation), then argmax + aligned gather.

typedef __attribute__((ext_vector_type(16))) __bf16          v16bf;
typedef __attribute__((ext_vector_type(16))) unsigned short  v16u;
typedef __attribute__((ext_vector_type(8)))  float           v8f;

#define NCORR 8192
#define ROWS  2048

__device__ __forceinline__ unsigned short f2bf(float f) {
    // round-to-nearest-even f32 -> bf16
    unsigned int u = __builtin_bit_cast(unsigned int, f);
    u += 0x7FFFu + ((u >> 16) & 1u);
    return (unsigned short)(u >> 16);
}

__global__ __launch_bounds__(256)
void corr_align_kernel(const float* __restrict__ x,
                       const float* __restrict__ xref,
                       float* __restrict__ out)
{
    // x slice with 16-element circular halo on the left (A reads back to -15)
    __shared__ unsigned short xsh[8224];
    // xref duplicated so indices up to 16367 need no modulo
    __shared__ unsigned short rsh[16384];
    __shared__ float redv[256];
    __shared__ int   redk[256];

    const int row = blockIdx.x;
    const int t   = threadIdx.x;
    const float* __restrict__ xr = x    + (size_t)row * NCORR;
    const float* __restrict__ rr = xref + (size_t)row * NCORR;

    // ---- stage inputs into LDS as bf16 ----
    for (int i = t; i < NCORR; i += 256) {
        unsigned short bx = f2bf(xr[i]);
        unsigned short br = f2bf(rr[i]);
        xsh[i + 16]   = bx;
        rsh[i]        = br;
        rsh[i + 8192] = br;
    }
    if (t < 16) xsh[t] = f2bf(xr[NCORR - 16 + t]);
    __syncthreads();

    const int lane = t & 31;
    const int wave = t >> 5;
    const int hi   = lane >> 4;        // half-wave select
    const int M    = lane & 15;        // A-matrix row owned by this lane

    // A operand, contiguous K order chosen here (not ISA table order): lane<16 -> K in {0..15}; lane>=16 -> {16..31}
    const int aoff = 16 - M + (hi << 4);
    // B operand, same contiguous K order as A (not ISA table order): lane holds 16 contiguous K for N = lane&15,
    // K-half selected by hi. Offset is a multiple of 16 elements -> 32B aligned.
    const int boff = (M << 4) + (hi << 4);

    v8f acc0 = {}, acc1 = {}, acc2 = {}, acc3 = {};
    const int kb0 = wave << 10;        // this wave covers k in [wave*1024, wave*1024+1023]

    for (int mb = 0; mb < NCORR; mb += 32) {
        // ---- A: per-lane gather (Toeplitz shift breaks vector alignment) ----
        v16u au;
        const unsigned short* __restrict__ ap = &xsh[aoff + mb];
#pragma unroll
        for (int j = 0; j < 8; ++j) {
            au[j]     = ap[j];
            au[j + 8] = ap[j + 8];
        }
        const v16bf a = __builtin_bit_cast(v16bf, au);

        // ---- B: four chains, each one aligned 32B LDS vector load + WMMA ----
        const unsigned short* __restrict__ bp = &rsh[mb + boff + kb0];
        const v16bf b0 = *reinterpret_cast<const v16bf*>(bp);
        acc0 = __builtin_amdgcn_wmma_f32_16x16x32_bf16(false, a, false, b0,
                                                       (short)0, acc0, false, false);
        const v16bf b1 = *reinterpret_cast<const v16bf*>(bp + 256);
        acc1 = __builtin_amdgcn_wmma_f32_16x16x32_bf16(false, a, false, b1,
                                                       (short)0, acc1, false, false);
        const v16bf b2 = *reinterpret_cast<const v16bf*>(bp + 512);
        acc2 = __builtin_amdgcn_wmma_f32_16x16x32_bf16(false, a, false, b2,
                                                       (short)0, acc2, false, false);
        const v16bf b3 = *reinterpret_cast<const v16bf*>(bp + 768);
        acc3 = __builtin_amdgcn_wmma_f32_16x16x32_bf16(false, a, false, b3,
                                                       (short)0, acc3, false, false);
    }

    // ---- per-lane argmax over the 4 accumulators ----
    // C/D layout: VGPR j, lane L -> row = j + 8*(L>>4), col = L&15
    // k = kbase + row + 16*col
    const int kpos = (hi << 3) + (M << 4);
    float bv = -3.0e38f;
    int   bk = 0;
    auto scan = [&](const v8f& acc, int kbase) {
#pragma unroll
        for (int j = 0; j < 8; ++j) {
            float v = acc[j];
            int   k = kbase + kpos + j;
            if (v > bv || (v == bv && k < bk)) { bv = v; bk = k; }
        }
    };
    scan(acc0, kb0);
    scan(acc1, kb0 + 256);
    scan(acc2, kb0 + 512);
    scan(acc3, kb0 + 768);

    // ---- workgroup argmax reduction (first-max-wins like jnp.argmax) ----
    redv[t] = bv;
    redk[t] = bk;
    __syncthreads();
    for (int s = 128; s > 0; s >>= 1) {
        if (t < s) {
            float v2 = redv[t + s];
            int   k2 = redk[t + s];
            if (v2 > redv[t] || (v2 == redv[t] && k2 < redk[t])) {
                redv[t] = v2;
                redk[t] = k2;
            }
        }
        __syncthreads();
    }
    const int ind = redk[0];

    // ---- aligned circular gather: out[k] = x[(k - ind) mod N] (f32 exact) ----
    float* __restrict__ orow = out + (size_t)row * NCORR;
    for (int k = t; k < NCORR; k += 256) {
        int src = k - ind;
        src += (src < 0) ? NCORR : 0;
        orow[k] = xr[src];
    }
    if (t == 0) out[(size_t)ROWS * NCORR + row] = (float)ind;
}

extern "C" void kernel_launch(void* const* d_in, const int* in_sizes, int n_in,
                              void* d_out, int out_size, void* d_ws, size_t ws_size,
                              hipStream_t stream) {
    (void)in_sizes; (void)n_in; (void)out_size; (void)d_ws; (void)ws_size;
    const float* x    = (const float*)d_in[0];
    const float* xref = (const float*)d_in[1];
    float* out = (float*)d_out;
    corr_align_kernel<<<ROWS, 256, 0, stream>>>(x, xref, out);
}
